// Scale_41523743817857
// MI455X (gfx1250) — matrix operand order repaired and checked by reading, not compiled, not run on hardware
//
#include <hip/hip_runtime.h>
#include <hip/hip_bf16.h>
#include <math.h>

// ---------------------------------------------------------------------------
// k_gemm:
//   - block = 8 waves, one predictor, 256 pixels (4 image rows)
//   - wave  = 32 pixels (two 16-row M-tiles sharing every B fragment)
//   - A: 6 padded ctx rows (contiguous 50.7 KB) -> LDS once via TDM
//   - B: K=64 tiles (128x64 bf16 = 16 KB) -> LDS, double-buffered via TDM
//   - B fragments register-double-buffered: ds_loads for tile t+1 are in
//     flight while the WMMA pair for tile t executes
// ---------------------------------------------------------------------------

typedef __attribute__((ext_vector_type(16))) __bf16 v16bf;
typedef __attribute__((ext_vector_type(8)))  float  v8f;
typedef __attribute__((ext_vector_type(4)))  unsigned int v4u;
typedef __attribute__((ext_vector_type(8)))  int v8i;
typedef __attribute__((ext_vector_type(4)))  int v4i;

union Frag32B { uint4 q[2]; v16bf v; };

static __device__ __forceinline__ unsigned short f32_to_bf16(float f) {
    union { float f; unsigned int u; } c; c.f = f;
    unsigned int u = c.u;
    u += 0x7FFFu + ((u >> 16) & 1u);      // round-to-nearest-even
    return (unsigned short)(u >> 16);
}

// ------------------------- small prep kernels ------------------------------

__global__ void k_zero_u32(unsigned int* p, int n) {
    int i = blockIdx.x * blockDim.x + threadIdx.x;
    if (i < n) p[i] = 0u;
}

__global__ void k_gamma(const float* __restrict__ ex, float* __restrict__ g, int n) {
    int i = blockIdx.x * blockDim.x + threadIdx.x;
    if (i < n) g[i] = powf(ex[i], 2.2f);
}

// t1[b,yo,x,c] = sum_j k[j] * g[b, 2*yo+j, x, c]   (8,128,266,3)
__global__ void k_convH(const float* __restrict__ g, float* __restrict__ t1, int n) {
    int i = blockIdx.x * blockDim.x + threadIdx.x;
    if (i >= n) return;
    int c  = i % 3;
    int x  = (i / 3) % 266;
    int yo = (i / 798) % 128;
    int b  = i / (798 * 128);
    float dot = 0.f, s = 0.f;
    const float PI = 3.14159265358979323846f;
    #pragma unroll
    for (int j = 0; j < 12; ++j) {
        float p  = (j - 5.5f) * 0.5f;
        float p3 = p * (1.0f / 3.0f);
        float kj = (sinf(PI * p) / (PI * p)) * (sinf(PI * p3) / (PI * p3));
        s += kj;
        dot += kj * g[((b * 266 + 2 * yo + j) * 266 + x) * 3 + c];
    }
    t1[i] = dot / s;
}

// small[b,ys,xs,c] -> ctx[b, ys/2 +1, xs/2 +1, (ys%2*2 + xs%2)*3 + c]
__global__ void k_convW(const float* __restrict__ t1, unsigned short* __restrict__ ctx, int n) {
    int i = blockIdx.x * blockDim.x + threadIdx.x;
    if (i >= n) return;
    int c  = i % 3;
    int xs = (i / 3) % 128;
    int ys = (i / 384) % 128;
    int b  = i / 49152;
    float dot = 0.f, s = 0.f;
    const float PI = 3.14159265358979323846f;
    #pragma unroll
    for (int j = 0; j < 12; ++j) {
        float p  = (j - 5.5f) * 0.5f;
        float p3 = p * (1.0f / 3.0f);
        float kj = (sinf(PI * p) / (PI * p)) * (sinf(PI * p3) / (PI * p3));
        s += kj;
        dot += kj * t1[((b * 128 + ys) * 266 + 2 * xs + j) * 3 + c];
    }
    float y = dot / s;
    float v = (y > 0.f) ? powf(fmaxf(y, 1e-20f), 1.0f / 2.2f) : 0.f;
    v = v * 2.f - 1.f;
    int Y = ys >> 1, X = xs >> 1;
    int ch = ((ys & 1) * 2 + (xs & 1)) * 3 + c;
    ctx[((b * 66 + Y + 1) * 66 + X + 1) * 64 + ch] = f32_to_bf16(v);
}

// example crop *2-1, space_to_depth(4) -> ctx channels 12..59
__global__ void k_ex(const float* __restrict__ ex, unsigned short* __restrict__ ctx, int n) {
    int i = blockIdx.x * blockDim.x + threadIdx.x;
    if (i >= n) return;
    int c  = i % 3;
    int rx = (i / 3) % 4;
    int ry = (i / 12) % 4;
    int X  = (i / 48) % 64;
    int Y  = (i / 3072) % 64;
    int b  = i / 196608;
    float v = ex[((b * 266 + 5 + 4 * Y + ry) * 266 + 5 + 4 * X + rx) * 3 + c] * 2.f - 1.f;
    ctx[((b * 66 + Y + 1) * 66 + X + 1) * 64 + 12 + (ry * 4 + rx) * 3 + c] = f32_to_bf16(v);
}

// W_conv (48,3,3,59,128) f32 -> Wp (48,128,576) bf16, zeroed past ch 12+i
__global__ void k_packW(const float* __restrict__ W, unsigned short* __restrict__ Wp, int n) {
    int i = blockIdx.x * blockDim.x + threadIdx.x;
    if (i >= n) return;
    int k  = i % 576;
    int nn = (i / 576) % 128;
    int p  = i / (576 * 128);           // predictor 0..47
    int ch = k % 64;
    int dx = (k / 64) % 3;
    int dy = k / 192;
    float v = 0.f;
    if (ch < 12 + p)                     // ch < 12+p <= 59, always in-bounds
        v = W[(((p * 3 + dy) * 3 + dx) * 59 + ch) * 128 + nn];
    Wp[i] = f32_to_bf16(v);
}

// --------------------------- TDM descriptor helper -------------------------
struct TdmDesc { v4u g0; v8i g1; };

static __device__ __forceinline__ TdmDesc tdm_2d(
    unsigned long long gaddr, unsigned lds_off,
    unsigned tdim0, unsigned tdim1,
    unsigned tile0, unsigned tile1,
    unsigned long long stride0)
{
    TdmDesc d;
    d.g0 = (v4u){ 1u,                                    // count=1
                  lds_off,                               // lds byte address
                  (unsigned)(gaddr & 0xFFFFFFFFu),
                  (unsigned)((gaddr >> 32) & 0x01FFFFFFu) | 0x80000000u }; // type=2
    d.g1 = (v8i){ (int)0x00010000u,                      // data_size = 2 bytes
                  (int)((tdim0 & 0xFFFFu) << 16),
                  (int)(((tdim0 >> 16) & 0xFFFFu) | ((tdim1 & 0xFFFFu) << 16)),
                  (int)(((tdim1 >> 16) & 0xFFFFu) | ((tile0 & 0xFFFFu) << 16)),
                  (int)(tile1 & 0xFFFFu),                // tile_dim1 (tile_dim2=0)
                  (int)(unsigned)(stride0 & 0xFFFFFFFFu),
                  (int)(unsigned)((stride0 >> 32) & 0xFFFFu),
                  0 };
    return d;
}

static __device__ __forceinline__ void tdm_issue(const TdmDesc& d) {
    __builtin_amdgcn_tensor_load_to_lds(d.g0, d.g1,
        (v4i){0,0,0,0}, (v4i){0,0,0,0},
        (v8i){0,0,0,0,0,0,0,0}, 0);
}

static __device__ __forceinline__ unsigned lds_offset(const void* p) {
    // LDS aperture: flat addr[31:0] == byte offset within group segment
    return (unsigned)(unsigned long long)p;
}

// --------------------------- main WMMA GEMM --------------------------------
__global__ __launch_bounds__(256) void k_gemm(
    const unsigned short* __restrict__ ctx,   // (8,66,66,64) bf16
    const unsigned short* __restrict__ Wp,    // (48,128,576) bf16
    const float* __restrict__ b_conv,         // (48,128)
    const float* __restrict__ W_dense,        // (48,128,1)
    const float* __restrict__ b_dense,        // (48,1)
    float* __restrict__ pred)                 // (8,64,64,48)
{
    __shared__ unsigned short sA[6 * 66 * 64];      // 25344 elems = 50688 B
    __shared__ unsigned short sB[2][128 * 64];      // 2 x 16384 B

    const int lane = threadIdx.x & 31;
    const int wave = threadIdx.x >> 5;
    const int i    = blockIdx.y;                    // predictor 0..47
    const int base = blockIdx.x * 256;              // first pixel of block
    const int b    = base >> 12;
    const int y0   = (base >> 6) & 63;              // multiple of 4
    const int rowInBuf = wave >> 1;                 // 0..3
    const int xbase = (wave & 1) * 32;              // 0 or 32

    const int m = lane & 15;
    const int h = lane >> 4;

    // ---- prologue: TDM A-chunk + B tile 0, wave 0 only ----
    if (wave == 0) {
        // A: 6 consecutive padded rows, fully contiguous: 6*66*64 elements
        unsigned long long ga =
            (unsigned long long)(ctx + ((size_t)(b * 66 + y0) * 66) * 64);
        TdmDesc da = tdm_2d(ga, lds_offset(sA),
                            25344u, 1u, 25344u, 1u, 25344ull);
        tdm_issue(da);
        // B tile 0: rows n=0..127 (stride 576), 64 contiguous k each
        unsigned long long gb =
            (unsigned long long)(Wp + (size_t)i * 128 * 576);
        TdmDesc db = tdm_2d(gb, lds_offset(sB[0]),
                            576u, 128u, 64u, 128u, 576ull);
        tdm_issue(db);
        __builtin_amdgcn_s_wait_tensorcnt(0);
    }
    __syncthreads();

    v8f acc0[8], acc1[8];
    #pragma unroll
    for (int t = 0; t < 8; ++t) {
        acc0[t] = (v8f){0.f,0.f,0.f,0.f,0.f,0.f,0.f,0.f};
        acc1[t] = (v8f){0.f,0.f,0.f,0.f,0.f,0.f,0.f,0.f};
    }

    for (int ss = 0; ss < 9; ++ss) {                // one 3x3 tap per ss
        // ---- kick off TDM for tile ss+1 into the other LDS buffer ----
        if (ss + 1 < 9 && wave == 0) {
            unsigned long long gb = (unsigned long long)
                (Wp + (size_t)i * 128 * 576 + (ss + 1) * 64);
            TdmDesc db = tdm_2d(gb, lds_offset(sB[(ss + 1) & 1]),
                                576u, 128u, 64u, 128u, 576ull);
            tdm_issue(db);
        }

        const int dy = ss / 3;
        const int dx = ss % 3;
        const unsigned short* sb = sB[ss & 1];

        #pragma unroll
        for (int half = 0; half < 2; ++half) {      // 32-wide K halves
            const int chalf = half * 32;

            // A fragments for both 16-pixel sub-tiles (16-bit A 16x32 layout)
            const int arow = (rowInBuf + dy) * 66;
            const int a0 = (arow + (xbase + m + dx)) * 64 + chalf + h * 8;
            const int a1 = (arow + (xbase + 16 + m + dx)) * 64 + chalf + h * 8;
            Frag32B af0, af1;
            af0.q[0] = *(const uint4*)(sA + a0);
            af0.q[1] = *(const uint4*)(sA + a0 + 16);
            af1.q[0] = *(const uint4*)(sA + a1);
            af1.q[1] = *(const uint4*)(sA + a1 + 16);

            // B fragments register-double-buffered: loads for tile t+1 are
            // outstanding while WMMA pair for tile t executes.
            const unsigned short* bbase = sb + chalf + h * 8 + m * 64;
            Frag32B bcur, bnext;
            bcur.q[0] = *(const uint4*)(bbase);
            bcur.q[1] = *(const uint4*)(bbase + 16);
            #pragma unroll
            for (int t = 0; t < 8; ++t) {
                if (t < 7) {
                    bnext.q[0] = *(const uint4*)(bbase + (t + 1) * 16 * 64);
                    bnext.q[1] = *(const uint4*)(bbase + (t + 1) * 16 * 64 + 16);
                }
                acc0[t] = __builtin_amdgcn_wmma_f32_16x16x32_bf16(
                    false, af0.v, false, bcur.v, (short)0, acc0[t], false, false);
                acc1[t] = __builtin_amdgcn_wmma_f32_16x16x32_bf16(
                    false, af1.v, false, bcur.v, (short)0, acc1[t], false, false);
                bcur = bnext;
            }
        }

        // ---- wait for next tile, publish to all waves ----
        if (ss + 1 < 9 && wave == 0)
            __builtin_amdgcn_s_wait_tensorcnt(0);
        __syncthreads();
    }

    // Epilogue: relu(acc + b_conv) . W_dense  (reduce across N=128)
    float pa0[8] = {0.f,0.f,0.f,0.f,0.f,0.f,0.f,0.f};
    float pa1[8] = {0.f,0.f,0.f,0.f,0.f,0.f,0.f,0.f};
    #pragma unroll
    for (int t = 0; t < 8; ++t) {
        const int n = t * 16 + m;
        const float bc = b_conv[i * 128 + n];
        const float wd = W_dense[i * 128 + n];
        #pragma unroll
        for (int r = 0; r < 8; ++r) {
            float v0 = fmaxf(acc0[t][r] + bc, 0.f);
            float v1 = fmaxf(acc1[t][r] + bc, 0.f);
            pa0[r] = fmaf(v0, wd, pa0[r]);
            pa1[r] = fmaf(v1, wd, pa1[r]);
        }
    }
    #pragma unroll
    for (int mask = 8; mask >= 1; mask >>= 1) {
        #pragma unroll
        for (int r = 0; r < 8; ++r) {
            pa0[r] += __shfl_xor(pa0[r], mask, 32);
            pa1[r] += __shfl_xor(pa1[r], mask, 32);
        }
    }
    if (m == 0) {
        const float bd = b_dense[i];
        const int y = y0 + rowInBuf;
        const int rowBase = ((b * 64 + y) * 64) * 48 + i;
        #pragma unroll
        for (int r = 0; r < 8; ++r) {
            // C/D layout: VGPR r -> M=r (lanes 0-15), M=8+r (lanes 16-31)
            const int x0 = xbase + h * 8 + r;
            const int x1 = xbase + 16 + h * 8 + r;
            pred[rowBase + x0 * 48] = pa0[r] + bd;
            pred[rowBase + x1 * 48] = pa1[r] + bd;
        }
    }
}

// depth_to_space(4) + *0.5+0.5 -> (8,256,256,3)
__global__ void k_out(const float* __restrict__ pred, float* __restrict__ out, int n) {
    int i = blockIdx.x * blockDim.x + threadIdx.x;
    if (i >= n) return;
    int c  = i % 3;
    int xo = (i / 3) % 256;
    int yo = (i / 768) % 256;
    int b  = i / 196608;
    int w = xo >> 2, rx = xo & 3;
    int hh = yo >> 2, ry = yo & 3;
    float v = pred[((b * 64 + hh) * 64 + w) * 48 + (ry * 4 + rx) * 3 + c];
    out[i] = v * 0.5f + 0.5f;
}

// ---------------------------------------------------------------------------

static inline size_t align256(size_t x) { return (x + 255) & ~(size_t)255; }

extern "C" void kernel_launch(void* const* d_in, const int* in_sizes, int n_in,
                              void* d_out, int out_size, void* d_ws, size_t ws_size,
                              hipStream_t stream) {
    const float* example = (const float*)d_in[0];   // (8,266,266,3)
    const float* W_conv  = (const float*)d_in[1];   // (48,3,3,59,128)
    const float* b_conv  = (const float*)d_in[2];   // (48,128)
    const float* W_dense = (const float*)d_in[3];   // (48,128,1)
    const float* b_dense = (const float*)d_in[4];   // (48,1)
    float* out = (float*)d_out;

    char* ws = (char*)d_ws;
    size_t cur = 0;
    const size_t g_bytes   = (size_t)8 * 266 * 266 * 3 * 4;   // 6.79 MB
    const size_t t1_bytes  = (size_t)8 * 128 * 266 * 3 * 4;   // 3.27 MB
    const size_t ctx_bytes = (size_t)8 * 66 * 66 * 64 * 2;    // 4.46 MB
    const size_t wp_bytes  = (size_t)48 * 128 * 576 * 2;      // 7.08 MB

    float* g = (float*)(ws + cur);                      // reused as pred later
    cur = align256(cur + g_bytes);
    float* t1 = (float*)(ws + cur);
    cur = align256(cur + t1_bytes);
    unsigned short* ctx = (unsigned short*)(ws + cur);
    cur = align256(cur + ctx_bytes);
    unsigned short* Wp = (unsigned short*)(ws + cur);
    cur = align256(cur + wp_bytes);
    float* pred = g;  // g is dead once t1 is built; alias to save workspace

    const int B = 256;
    {   // 1. zero ctx (halo + channels 60..63 stay zero)
        int n = (int)(ctx_bytes / 4);
        k_zero_u32<<<(n + B - 1) / B, B, 0, stream>>>((unsigned int*)ctx, n);
    }
    {   // 2. gamma expand
        int n = 8 * 266 * 266 * 3;
        k_gamma<<<(n + B - 1) / B, B, 0, stream>>>(example, g, n);
    }
    {   // 3. vertical sinc conv (stride 2)
        int n = 8 * 128 * 266 * 3;
        k_convH<<<(n + B - 1) / B, B, 0, stream>>>(g, t1, n);
    }
    {   // 4. horizontal sinc conv + degamma + s2d(2) -> ctx[0..11]
        int n = 8 * 128 * 128 * 3;
        k_convW<<<(n + B - 1) / B, B, 0, stream>>>(t1, ctx, n);
    }
    {   // 5. example crop + s2d(4) -> ctx[12..59]
        int n = 8 * 64 * 64 * 48;
        k_ex<<<(n + B - 1) / B, B, 0, stream>>>(example, ctx, n);
    }
    {   // 6. pack + zero-pad weights -> (48,128,576) bf16
        int n = 48 * 128 * 576;
        k_packW<<<(n + B - 1) / B, B, 0, stream>>>(W_conv, Wp, n);
    }
    {   // 7. the 48 fused conv-GEMMs (WMMA + TDM double buffering)
        dim3 grid(128, 48);
        k_gemm<<<grid, 256, 0, stream>>>(ctx, Wp, b_conv, W_dense, b_dense, pred);
    }
    {   // 8. depth_to_space + affine
        int n = 8 * 256 * 256 * 3;
        k_out<<<(n + B - 1) / B, B, 0, stream>>>(pred, out, n);
    }
    (void)in_sizes; (void)n_in; (void)out_size; (void)ws_size;
}
